// SWAttention_3908420239939
// MI455X (gfx1250) — matrix operand order repaired and checked by reading, not compiled, not run on hardware
//
#include <hip/hip_runtime.h>
#include <hip/hip_bf16.h>

typedef __attribute__((ext_vector_type(16))) _Float16 v16h;
typedef __attribute__((ext_vector_type(4)))  _Float16 v4h;
typedef __attribute__((ext_vector_type(8)))  float    v8f;

#define WMMA16(a, b, c) \
    __builtin_amdgcn_wmma_f32_16x16x32_f16(false, (a), false, (b), (short)0, (c), false, false)

// ---------------------------------------------------------------------------
// Fragment loaders (wave32, 16x16x32 f16 WMMA, contiguous k = 16*(lane/16) + elem for A and B)
// ---------------------------------------------------------------------------

// A-matrix 16x32 fragment from a row-major [rows x ld] array.
// lanes 0-15: M=lane, elems 0..15 = K0..15
// lanes16-31: M=lane-16, elems 0..15 = K16..31
__device__ __forceinline__ v16h load_a16(const _Float16* base, int ld, int lane) {
    const int r  = lane & 15;
    const int kh = (lane >> 4) * 16;
    const _Float16* p = base + r * ld + kh;
    v16h a;
#pragma unroll
    for (int i = 0; i < 8; ++i) a[i] = p[i];
#pragma unroll
    for (int i = 0; i < 8; ++i) a[8 + i] = p[8 + i];
    return a;
}

// B-matrix 32x16 fragment where source is laid out [n][k] row-major with
// leading dim ld (per-lane K run contiguous -> 2x b128 loads).
// lanes 0-15: N=lane, K=0..15 ; lanes 16-31: N=lane-16, K=16..31
__device__ __forceinline__ v16h load_b_rowk(const _Float16* base, int ld, int lane) {
    const int n  = lane & 15;
    const int k0 = (lane >> 4) * 16;
    const _Float16* p = base + n * ld + k0;
    v16h b;
#pragma unroll
    for (int i = 0; i < 16; ++i) b[i] = p[i];
    return b;
}

// ---------------------------------------------------------------------------
// 64x128 = (64x128 LDS X) @ (128x128 global W_h^T) + bias, result -> LDS.
// TR=false: D is 64x128 row-major.  TR=true: D is the 128x64 transpose.
// ---------------------------------------------------------------------------
template <bool TR>
__device__ __forceinline__ void gemm_x_w(const _Float16* Xs, const _Float16* W,
                                         const float* bias, _Float16* D, int lane) {
    for (int mt = 0; mt < 4; ++mt) {
        v16h a[4];
#pragma unroll
        for (int ks = 0; ks < 4; ++ks) a[ks] = load_a16(Xs + mt * 2048 + ks * 32, 128, lane);
#pragma unroll
        for (int nt = 0; nt < 8; ++nt) {
            const float bv = bias[nt * 16 + (lane & 15)];
            v8f acc;
#pragma unroll
            for (int j = 0; j < 8; ++j) acc[j] = bv;
#pragma unroll
            for (int ks = 0; ks < 4; ++ks)
                acc = WMMA16(a[ks], load_b_rowk(W + nt * 16 * 128 + ks * 32, 128, lane), acc);
            const int r0 = mt * 16 + ((lane >> 4) << 3);
            const int cc = nt * 16 + (lane & 15);
#pragma unroll
            for (int j = 0; j < 8; ++j) {
                if (TR) D[cc * 64 + r0 + j]      = (_Float16)acc[j];
                else    D[(r0 + j) * 128 + cc]   = (_Float16)acc[j];
            }
        }
    }
}

// ---------------------------------------------------------------------------
// Prep kernels
// ---------------------------------------------------------------------------
__global__ void cvt_f32_f16_kernel(const float* __restrict__ s, _Float16* __restrict__ d, int n) {
    int i = blockIdx.x * blockDim.x + threadIdx.x;
    if (i < n) d[i] = (_Float16)s[i];
}

__global__ void init_out_kernel(float* __restrict__ out, const float* __restrict__ pb, int n) {
    int i = blockIdx.x * blockDim.x + threadIdx.x;
    if (i < n) out[i] = pb[i & 127];
}

// ---------------------------------------------------------------------------
// Main kernel: one wave32 per (shifted-window, head). 8192 blocks total.
// ---------------------------------------------------------------------------
__global__ __launch_bounds__(32)
void swattn_kernel(const float* __restrict__ x,
                   const float* __restrict__ mask,   // (3,64,64)
                   const float* __restrict__ Bp,     // (64,64)
                   const float* __restrict__ qb,     // (1024,)
                   const float* __restrict__ kb,
                   const float* __restrict__ vb,
                   const _Float16* __restrict__ wq,  // (1024,128) f16
                   const _Float16* __restrict__ wk,
                   const _Float16* __restrict__ wv,
                   const _Float16* __restrict__ pw,  // (128,1024) f16
                   float* __restrict__ out)          // pre-initialized with proj bias
{
    __shared__ _Float16 smem[32768];                 // 64 KB
    _Float16* X  = smem;                             // 64x128  (dead after QKV)
    _Float16* Q  = smem + 8192;                      // 64x128  (dead after att)
    _Float16* Km = smem + 16384;                     // 64x128
    _Float16* Vt = smem + 24576;                     // 128x64  (V transposed)
    _Float16* Pm = smem;                             // 64x64   (aliases X)
    _Float16* O  = smem + 8192;                      // 64x128  (aliases Q)

    const int lane = threadIdx.x;
    const int wid  = blockIdx.x;
    const int h    = wid & 7;
    const int win  = wid >> 3;
    const int img  = win >> 8;
    const int wic  = win & 255;
    const int swy  = wic >> 4, swx = wic & 15;

    // ---- 1. gather shifted window (roll by -4 == read (y+4)%128), f32 -> f16 LDS
    for (int t = 0; t < 64; ++t) {
        const int sy = swy * 8 + (t >> 3), sx = swx * 8 + (t & 7);
        const int gy = (sy + 4) & 127,     gx = (sx + 4) & 127;
        const int src = ((img * 256 + (gy >> 3) * 16 + (gx >> 3)) * 64 +
                         ((gy & 7) * 8 + (gx & 7))) * 128;
        const float4 f = ((const float4*)(x + src))[lane];
        v4h hv;
        hv[0] = (_Float16)f.x; hv[1] = (_Float16)f.y;
        hv[2] = (_Float16)f.z; hv[3] = (_Float16)f.w;
        *(v4h*)(X + t * 128 + lane * 4) = hv;        // one ds_store_b64
    }
    __syncthreads();

    // ---- 2. Q, K, V = X @ W_h^T + b_h   (V stored transposed)
    gemm_x_w<false>(X, wq + h * 128 * 128, qb + h * 128, Q,  lane);
    gemm_x_w<false>(X, wk + h * 128 * 128, kb + h * 128, Km, lane);
    gemm_x_w<true >(X, wv + h * 128 * 128, vb + h * 128, Vt, lane);
    __syncthreads();

    // ---- 3. att = Q K^T / sqrt(128) + Bp  (64x64 held in 16 f32 fragments)
    const float inv_sdk = 0.08838834764831845f;
    v8f att[4][4];
    for (int mt = 0; mt < 4; ++mt) {
        v16h a[4];
#pragma unroll
        for (int ks = 0; ks < 4; ++ks) a[ks] = load_a16(Q + mt * 2048 + ks * 32, 128, lane);
#pragma unroll
        for (int nt = 0; nt < 4; ++nt) {
            v8f acc = {};
#pragma unroll
            for (int ks = 0; ks < 4; ++ks)
                acc = WMMA16(a[ks], load_b_rowk(Km + nt * 16 * 128 + ks * 32, 128, lane), acc);
            const int r0 = mt * 16 + ((lane >> 4) << 3);
            const int cc = nt * 16 + (lane & 15);
#pragma unroll
            for (int j = 0; j < 8; ++j) acc[j] = acc[j] * inv_sdk + Bp[(r0 + j) * 64 + cc];
            att[mt][nt] = acc;
        }
    }

    // ---- 4. row softmax (row = 16-lane half for fixed (mt, j))
    for (int mt = 0; mt < 4; ++mt) {
#pragma unroll
        for (int j = 0; j < 8; ++j) {
            float m = att[mt][0][j];
#pragma unroll
            for (int nt = 1; nt < 4; ++nt) m = fmaxf(m, att[mt][nt][j]);
#pragma unroll
            for (int off = 8; off >= 1; off >>= 1) m = fmaxf(m, __shfl_xor(m, off, 16));
            float s = 0.f;
#pragma unroll
            for (int nt = 0; nt < 4; ++nt) {
                const float e = __expf(att[mt][nt][j] - m);
                att[mt][nt][j] = e;
                s += e;
            }
#pragma unroll
            for (int off = 8; off >= 1; off >>= 1) s += __shfl_xor(s, off, 16);
            const float r = 1.0f / s;
#pragma unroll
            for (int nt = 0; nt < 4; ++nt) att[mt][nt][j] *= r;
        }
    }

    // ---- 5. border-window multiplicative mask (post-softmax)
    const int msel = (swy == 15 && swx == 15) ? 2 : (swy == 15 ? 0 : (swx == 15 ? 1 : -1));
    if (msel >= 0) {
        const float* M = mask + msel * 64 * 64;
        for (int mt = 0; mt < 4; ++mt)
#pragma unroll
            for (int nt = 0; nt < 4; ++nt) {
                const int r0 = mt * 16 + ((lane >> 4) << 3);
                const int cc = nt * 16 + (lane & 15);
#pragma unroll
                for (int j = 0; j < 8; ++j) att[mt][nt][j] *= M[(r0 + j) * 64 + cc];
            }
    }

    // ---- 6. P -> LDS f16 (reuse X region)
    __syncthreads();
    for (int mt = 0; mt < 4; ++mt)
#pragma unroll
        for (int nt = 0; nt < 4; ++nt) {
            const int r0 = mt * 16 + ((lane >> 4) << 3);
            const int cc = nt * 16 + (lane & 15);
#pragma unroll
            for (int j = 0; j < 8; ++j) Pm[(r0 + j) * 64 + cc] = (_Float16)att[mt][nt][j];
        }
    __syncthreads();

    // ---- 7. O = P (64x64) @ V (64x128) -> LDS f16 (reuse Q region)
    //      B[k][n] = V[k][n] = Vt[n][k]  -> contiguous per-lane K run in Vt
    for (int mt = 0; mt < 4; ++mt) {
        v16h a[2];
#pragma unroll
        for (int ks = 0; ks < 2; ++ks) a[ks] = load_a16(Pm + mt * 1024 + ks * 32, 64, lane);
#pragma unroll
        for (int nt = 0; nt < 8; ++nt) {
            v8f acc = {};
#pragma unroll
            for (int ks = 0; ks < 2; ++ks)
                acc = WMMA16(a[ks], load_b_rowk(Vt + nt * 16 * 64 + ks * 32, 64, lane), acc);
            const int r0 = mt * 16 + ((lane >> 4) << 3);
            const int cc = nt * 16 + (lane & 15);
#pragma unroll
            for (int j = 0; j < 8; ++j) O[(r0 + j) * 128 + cc] = (_Float16)acc[j];
        }
    }
    __syncthreads();

    // ---- 8. per-head projection slice + atomic scatter with reverse roll (+4)
    for (int mt = 0; mt < 4; ++mt) {
        v16h a[4];
#pragma unroll
        for (int ks = 0; ks < 4; ++ks) a[ks] = load_a16(O + mt * 2048 + ks * 32, 128, lane);
#pragma unroll
        for (int ct = 0; ct < 8; ++ct) {
            v8f acc = {};
#pragma unroll
            for (int ks = 0; ks < 4; ++ks)
                acc = WMMA16(a[ks], load_b_rowk(pw + ct * 16 * 1024 + h * 128 + ks * 32, 1024, lane), acc);
            const int r0 = mt * 16 + ((lane >> 4) << 3);
            const int cc = ct * 16 + (lane & 15);
#pragma unroll
            for (int j = 0; j < 8; ++j) {
                const int t  = r0 + j;
                const int sy = swy * 8 + (t >> 3), sx = swx * 8 + (t & 7);
                const int fy = (sy + 4) & 127,     fx = (sx + 4) & 127;
                const int dst = ((img * 256 + (fy >> 3) * 16 + (fx >> 3)) * 64 +
                                 ((fy & 7) * 8 + (fx & 7))) * 128 + cc;
                atomicAdd(out + dst, acc[j]);
            }
        }
    }
}

// ---------------------------------------------------------------------------
// Launch
// ---------------------------------------------------------------------------
extern "C" void kernel_launch(void* const* d_in, const int* in_sizes, int n_in,
                              void* d_out, int out_size, void* d_ws, size_t ws_size,
                              hipStream_t stream) {
    const float* x      = (const float*)d_in[0];
    const float* mask   = (const float*)d_in[1];
    const float* Bp     = (const float*)d_in[2];
    const float* wq_w   = (const float*)d_in[3];
    const float* wq_b   = (const float*)d_in[4];
    const float* wk_w   = (const float*)d_in[5];
    const float* wk_b   = (const float*)d_in[6];
    const float* wv_w   = (const float*)d_in[7];
    const float* wv_b   = (const float*)d_in[8];
    const float* proj_w = (const float*)d_in[9];
    const float* proj_b = (const float*)d_in[10];
    float* outp = (float*)d_out;

    const int WELEM = 1024 * 128;                 // 131072 per weight matrix
    _Float16* wq_h = (_Float16*)d_ws;
    _Float16* wk_h = wq_h + WELEM;
    _Float16* wv_h = wk_h + WELEM;
    _Float16* pw_h = wv_h + WELEM;

    cvt_f32_f16_kernel<<<(WELEM + 255) / 256, 256, 0, stream>>>(wq_w, wq_h, WELEM);
    cvt_f32_f16_kernel<<<(WELEM + 255) / 256, 256, 0, stream>>>(wk_w, wk_h, WELEM);
    cvt_f32_f16_kernel<<<(WELEM + 255) / 256, 256, 0, stream>>>(wv_w, wv_h, WELEM);
    cvt_f32_f16_kernel<<<(WELEM + 255) / 256, 256, 0, stream>>>(proj_w, pw_h, WELEM);

    init_out_kernel<<<(out_size + 255) / 256, 256, 0, stream>>>(outp, proj_b, out_size);

    const int nwin = in_sizes[0] / (64 * 128);    // total windows (1024)
    swattn_kernel<<<nwin * 8, 32, 0, stream>>>(x, mask, Bp, wq_b, wk_b, wv_b,
                                               wq_h, wk_h, wv_h, pw_h, outp);
}
